// Encoder_Block_7112465842259
// MI455X (gfx1250) — matrix operand order repaired and checked by reading, not compiled, not run on hardware
//
#include <hip/hip_runtime.h>
#include <math.h>

typedef __attribute__((ext_vector_type(16))) _Float16 v16h;
typedef __attribute__((ext_vector_type(8)))  float    v8f;

#define BN_EPS 1e-5f

// ---------------------------------------------------------------------------
// Weight pre-pack: OIHW f32 -> f16 in the contiguous-k B-fragment layout this file uses
//   wp[((nt*NCHUNK + kc)*32 + lane)*16 + h] = w[co=nt*16+lane%16][k=kc*32+h+16*(lane/16)]
// so each lane's 16-half B fragment is one contiguous, 32B-aligned vector.
// ---------------------------------------------------------------------------
__global__ __launch_bounds__(256)
void pack_w_kernel(const float* __restrict__ w, _Float16* __restrict__ wp,
                   int Cout, int Ktot)
{
    int idx = blockIdx.x * blockDim.x + threadIdx.x;    // Cout*Ktot total
    if (idx >= Cout * Ktot) return;
    int h    = idx & 15;
    int lane = (idx >> 4) & 31;
    int nchunk = Ktot >> 5;
    int kc = (idx >> 9) % nchunk;
    int nt = (idx >> 9) / nchunk;
    int lh = lane >> 4, lm = lane & 15;
    int co = nt * 16 + lm;
    int k  = kc * 32 + h + lh * 16;
    wp[idx] = (_Float16)w[(size_t)co * Ktot + k];
}

// ---------------------------------------------------------------------------
// Implicit-GEMM conv via V_WMMA_F32_16X16X32_F16, fully templated (incl. the
// Cout/64 group count NG) so all index math is compile-time shifts/masks.
// One wave computes a 16(M=pixels) x 64(N=4 co-tiles) block: the im2col A
// fragment is gathered once per K-chunk and reused by 4 WMMAs.
// Output is always 128x128 spatial here (WOUT=128 -> shifts).
// Epilogue folds bias + eval-BN + ReLU.
// ---------------------------------------------------------------------------
template <int CIN, int HIN, int WIN, int KH, int KW, int STRIDE, int PAD, int NG>
__global__ __launch_bounds__(32)
void conv_wmma_kernel(const float* __restrict__ in, const _Float16* __restrict__ wp,
                      const float* __restrict__ bias,
                      const float* __restrict__ gamma, const float* __restrict__ beta,
                      float* __restrict__ out,
                      int CoutTotal, int coOffset)
{
    constexpr int KK     = KH * KW;
    constexpr int KTOT   = CIN * KK;       // multiple of 32
    constexpr int NCHUNK = KTOT / 32;
    constexpr int WOUT   = 128;            // all convs here emit 128x128

    const int lane = threadIdx.x;
    const int lh   = lane >> 4;            // lane half: 0 or 1
    const int lm   = lane & 15;

    const int mtile = blockIdx.x / NG;     // NG is 1 or 2 -> shift
    const int grp   = blockIdx.x - mtile * NG;
    const int b     = blockIdx.y;

    // A row for this lane: output pixel index
    const int mA  = mtile * 16 + lm;
    const int oyA = mA >> 7;
    const int oxA = mA & (WOUT - 1);
    const float* __restrict__ inB = in + (size_t)b * CIN * HIN * WIN;

    const _Float16* __restrict__ wg = wp + (size_t)(grp * 4) * NCHUNK * 512;

    v8f acc[4];
#pragma unroll
    for (int j = 0; j < 4; ++j) acc[j] = (v8f){0.f,0.f,0.f,0.f,0.f,0.f,0.f,0.f};

    for (int kc = 0; kc < NCHUNK; ++kc) {
        const int kk = kc * 32;
        // ---- gather im2col A fragment (16x32 f16 layout) ----
        v16h afrag;
#pragma unroll
        for (int h = 0; h < 16; ++h) {
            // lanes 0-15 -> K = kk+0..15; lanes 16-31 -> kk+16..31
            int kA  = kk + h + lh * 16;
            int cin = kA / KK;                 // KK is constexpr -> cheap
            int r   = kA - cin * KK;
            int ky  = r / KW;
            int kx  = r - ky * KW;
            int iy  = oyA * STRIDE - PAD + ky;
            int ix  = oxA * STRIDE - PAD + kx;
            float va = 0.f;
            if (iy >= 0 && iy < HIN && ix >= 0 && ix < WIN)
                va = inB[((size_t)cin * HIN + iy) * WIN + ix];
            afrag[h] = (_Float16)va;
        }
        // prefetch next K-chunk of packed weights (global_prefetch_b8)
        if (kc + 1 < NCHUNK)
            __builtin_prefetch(wg + (size_t)(kc + 1) * 512 + lane * 16, 0, 1);
        // ---- 4 WMMAs reuse the A fragment across 64 output channels ----
#pragma unroll
        for (int j = 0; j < 4; ++j) {
            const v16h bfrag =
                *(const v16h*)(wg + ((size_t)(j * NCHUNK + kc) * 32 + lane) * 16);
            acc[j] = __builtin_amdgcn_wmma_f32_16x16x32_f16(
                false, afrag, false, bfrag, (short)0, acc[j], false, false);
        }
    }

    // ---- epilogue: bias + BN + ReLU, D layout: VGPR r -> M=r+8*lh, N=lm ----
#pragma unroll
    for (int j = 0; j < 4; ++j) {
        const int co = (grp * 4 + j) * 16 + lm;
        const float bi = bias  ? bias[co]  : 0.f;
        const float s  = gamma ? gamma[co] * rsqrtf(1.f + BN_EPS) : 1.f;
        const float bb = beta  ? beta[co]  : 0.f;
        float* orow = out + ((size_t)b * CoutTotal + coOffset + co) * 16384;
#pragma unroll
        for (int r = 0; r < 8; ++r) {
            int m = mtile * 16 + r + 8 * lh;   // pixel index (oy*128+ox == m)
            float v = (acc[j][r] + bi) * s + bb;
            v = v > 0.f ? v : 0.f;
            orow[m] = v;
        }
    }
}

// ---------------------------------------------------------------------------
// Downsample pools: even channels -> maxpool2, odd channels -> avgpool2,
// written to out channels 64..127 of the 128-channel buffer.
// ---------------------------------------------------------------------------
__global__ __launch_bounds__(256)
void pool_kernel(const float* __restrict__ x, float* __restrict__ out)
{
    int idx = blockIdx.x * blockDim.x + threadIdx.x;     // 4*64*16384 total
    int p = idx & 16383;
    int c = (idx >> 14) & 63;
    int b = idx >> 20;
    int oy = p >> 7, ox = p & 127;
    int cin; bool ismax;
    if (c < 32) { cin = 2 * c;            ismax = true;  }
    else        { cin = 2 * (c - 32) + 1; ismax = false; }
    const float* src = x + (((size_t)b * 64 + cin) * 256 + oy * 2) * 256 + ox * 2;
    float v00 = src[0], v01 = src[1], v10 = src[256], v11 = src[257];
    float v = ismax ? fmaxf(fmaxf(v00, v01), fmaxf(v10, v11))
                    : 0.25f * (v00 + v01 + v10 + v11);
    out[(((size_t)b * 128 + 64 + c) << 14) + p] = v;
}

// ---------------------------------------------------------------------------
// Depthwise 3x3 (pad 1) on the EVEN channels of a 128-channel input.
// ---------------------------------------------------------------------------
__global__ __launch_bounds__(256)
void dw_kernel(const float* __restrict__ x, const float* __restrict__ w,
               const float* __restrict__ bias, float* __restrict__ out)
{
    int idx = blockIdx.x * blockDim.x + threadIdx.x;     // 4*64*16384 total
    int p = idx & 16383;
    int j = (idx >> 14) & 63;
    int b = idx >> 20;
    int oy = p >> 7, ox = p & 127;
    const float* src = x + (((size_t)b * 128 + 2 * j) << 14);
    const float* wk  = w + j * 9;
    float acc = bias[j];
#pragma unroll
    for (int ky = 0; ky < 3; ++ky)
#pragma unroll
        for (int kx = 0; kx < 3; ++kx) {
            int iy = oy - 1 + ky, ix = ox - 1 + kx;
            if (iy >= 0 && iy < 128 && ix >= 0 && ix < 128)
                acc += src[(iy << 7) + ix] * wk[ky * 3 + kx];
        }
    out[(((size_t)b * 64 + j) << 14) + p] = acc;
}

// ---------------------------------------------------------------------------
// Multiscale cyclic conv on ODD channels: channel j uses dilation (j%4)+1,
// weight mcc_w[j%4], then BN(bn[64+j]) + ReLU into out channels 64..127.
// ---------------------------------------------------------------------------
__global__ __launch_bounds__(256)
void mcc_kernel(const float* __restrict__ x, const float* __restrict__ mccw,
                const float* __restrict__ mccb,
                const float* __restrict__ bng, const float* __restrict__ bnb,
                float* __restrict__ out)
{
    int idx = blockIdx.x * blockDim.x + threadIdx.x;     // 4*64*16384 total
    int p = idx & 16383;
    int j = (idx >> 14) & 63;
    int b = idx >> 20;
    int oy = p >> 7, ox = p & 127;
    int d = j & 3;
    int dil = d + 1;
    const float* src = x + (((size_t)b * 128 + 2 * j + 1) << 14);
    const float* wk  = mccw + d * 9;
    float acc = mccb[d];
#pragma unroll
    for (int ky = 0; ky < 3; ++ky)
#pragma unroll
        for (int kx = 0; kx < 3; ++kx) {
            int iy = oy + (ky - 1) * dil, ix = ox + (kx - 1) * dil;
            if (iy >= 0 && iy < 128 && ix >= 0 && ix < 128)
                acc += src[(iy << 7) + ix] * wk[ky * 3 + kx];
        }
    float s = bng[64 + j] * rsqrtf(1.f + BN_EPS);
    float v = acc * s + bnb[64 + j];
    v = v > 0.f ? v : 0.f;
    out[(((size_t)b * 128 + 64 + j) << 14) + p] = v;
}

// ---------------------------------------------------------------------------
// Per-(b,c) mean & max over 128x128 pixels. One block per (b,c).
// ---------------------------------------------------------------------------
__global__ __launch_bounds__(256)
void chanstat_kernel(const float* __restrict__ x,
                     float* __restrict__ avgOut, float* __restrict__ maxOut)
{
    __shared__ float ssum[256];
    __shared__ float smax[256];
    int bc = blockIdx.x;
    int tid = threadIdx.x;
    const float* src = x + ((size_t)bc << 14);
    float s = 0.f, m = -INFINITY;
    for (int i = tid; i < 16384; i += 256) {
        float v = src[i];
        s += v;
        m = fmaxf(m, v);
    }
    ssum[tid] = s; smax[tid] = m;
    __syncthreads();
    for (int off = 128; off > 0; off >>= 1) {
        if (tid < off) {
            ssum[tid] += ssum[tid + off];
            smax[tid] = fmaxf(smax[tid], smax[tid + off]);
        }
        __syncthreads();
    }
    if (tid == 0) {
        avgOut[bc] = ssum[0] * (1.f / 16384.f);
        maxOut[bc] = smax[0];
    }
}

// ECA conv1d (k=3, pad=1) over channel axis on stacked [max;avg], sigmoid.
__global__ __launch_bounds__(256)
void ca_kernel(const float* __restrict__ avg, const float* __restrict__ mx,
               const float* __restrict__ w, const float* __restrict__ bias,
               float* __restrict__ ca)
{
    int idx = blockIdx.x * blockDim.x + threadIdx.x;
    if (idx >= 512) return;                              // 4*128
    int c = idx & 127, b = idx >> 7;
    const float* A = avg + b * 128;
    const float* M = mx + b * 128;
    float acc = bias[0];
#pragma unroll
    for (int j = 0; j < 3; ++j) {
        int cc = c + j - 1;
        if (cc >= 0 && cc < 128)
            acc += w[j] * M[cc] + w[3 + j] * A[cc];      // ch0 = max, ch1 = avg
    }
    ca[idx] = 1.f / (1.f + expf(-acc));
}

// Per-(b,pixel) channel max/mean of residual*ca (spatial attention stats).
__global__ __launch_bounds__(256)
void sstat_kernel(const float* __restrict__ res, const float* __restrict__ ca,
                  float* __restrict__ smax, float* __restrict__ savg)
{
    int idx = blockIdx.x * blockDim.x + threadIdx.x;     // 4*16384 total
    int p = idx & 16383, b = idx >> 14;
    float m = -INFINITY, s = 0.f;
    for (int c = 0; c < 128; ++c) {
        float v = res[(((size_t)b * 128 + c) << 14) + p] * ca[(b << 7) + c];
        m = fmaxf(m, v);
        s += v;
    }
    smax[idx] = m;
    savg[idx] = s * (1.f / 128.f);
}

// Spatial attention: 7x7 conv (pad 3) over [smax;savg], sigmoid.
__global__ __launch_bounds__(256)
void sa_kernel(const float* __restrict__ smax, const float* __restrict__ savg,
               const float* __restrict__ w, const float* __restrict__ bias,
               float* __restrict__ sa)
{
    int idx = blockIdx.x * blockDim.x + threadIdx.x;     // 4*16384 total
    int p = idx & 16383, b = idx >> 14;
    int oy = p >> 7, ox = p & 127;
    float acc = bias[0];
#pragma unroll
    for (int ky = 0; ky < 7; ++ky)
#pragma unroll
        for (int kx = 0; kx < 7; ++kx) {
            int iy = oy - 3 + ky, ix = ox - 3 + kx;
            if (iy >= 0 && iy < 128 && ix >= 0 && ix < 128) {
                int q = (b << 14) + (iy << 7) + ix;
                acc += w[ky * 7 + kx] * smax[q] + w[49 + ky * 7 + kx] * savg[q];
            }
        }
    sa[idx] = 1.f / (1.f + expf(-acc));
}

// out = residual + (residual*ca)*sa  ==  residual * (1 + ca*sa), in place.
__global__ __launch_bounds__(256)
void final_kernel(float* __restrict__ out, const float* __restrict__ ca,
                  const float* __restrict__ sa)
{
    int idx = blockIdx.x * blockDim.x + threadIdx.x;     // 4*128*16384 total
    int p = idx & 16383;
    int c = (idx >> 14) & 127;
    int b = idx >> 21;
    out[idx] = out[idx] * (1.f + ca[(b << 7) + c] * sa[(b << 14) + p]);
}

// ---------------------------------------------------------------------------
// basic_conv block: depthwise + WMMA pointwise on even channels,
// multiscale dilated convs on odd channels, BN+ReLU fused into both paths.
// ---------------------------------------------------------------------------
static void run_basic(const float* in, float* out, void* const* pp,
                      float* tmp_dw, _Float16* wp_pw, hipStream_t stream)
{
    const float* dw_w  = (const float*)pp[0];
    const float* dw_b  = (const float*)pp[1];
    const float* pw_b  = (const float*)pp[3];
    const float* mcc_w = (const float*)pp[4];
    const float* mcc_b = (const float*)pp[5];
    const float* bn_g  = (const float*)pp[6];
    const float* bn_b  = (const float*)pp[7];

    dw_kernel<<<(4 * 64 * 16384) / 256, 256, 0, stream>>>(in, dw_w, dw_b, tmp_dw);
    // pointwise 64->64 (K=64) + bias + BN(ch 0..63) + ReLU -> out channels 0..63
    conv_wmma_kernel<64, 128, 128, 1, 1, 1, 0, 1>
        <<<dim3(16384 / 16, 4), 32, 0, stream>>>(
            tmp_dw, wp_pw, pw_b, bn_g, bn_b, out, 128, 0);
    mcc_kernel<<<(4 * 64 * 16384) / 256, 256, 0, stream>>>(in, mcc_w, mcc_b,
                                                           bn_g, bn_b, out);
}

extern "C" void kernel_launch(void* const* d_in, const int* in_sizes, int n_in,
                              void* d_out, int out_size, void* d_ws, size_t ws_size,
                              hipStream_t stream)
{
    // setup_inputs() flattened order:
    // 0:x 1:ds_w 2:ds_g 3:ds_b 4..11:p1{dw_w,dw_b,pw_w,pw_b,mcc_w,mcc_b,bn_g,bn_b}
    // 12..19:p2 20:cbr_w 21:cbr_b 22:cbr_g 23:cbr_beta 24:ca_w 25:ca_b 26:sa_w 27:sa_b
    const float* x        = (const float*)d_in[0];
    const float* ds_w     = (const float*)d_in[1];
    const float* ds_g     = (const float*)d_in[2];
    const float* ds_b     = (const float*)d_in[3];
    const float* p1_pw_w  = (const float*)d_in[6];
    const float* p2_pw_w  = (const float*)d_in[14];
    const float* cbr_w    = (const float*)d_in[20];
    const float* cbr_b    = (const float*)d_in[21];
    const float* cbr_g    = (const float*)d_in[22];
    const float* cbr_beta = (const float*)d_in[23];
    const float* ca_w     = (const float*)d_in[24];
    const float* ca_b     = (const float*)d_in[25];
    const float* sa_w     = (const float*)d_in[26];
    const float* sa_b     = (const float*)d_in[27];

    float* ws   = (float*)d_ws;
    float* buf0 = ws;                       // [4,128,128,128] ds out / bc2 out
    float* buf2 = ws + 8388608;             // [4,128,128,128] bc1 out
    float* buf1 = ws + 16777216;            // [4,64,128,128]  depthwise tmp
    float* st   = ws + 20971520;
    float* chAvg = st;                      // 512
    float* chMax = st + 512;                // 512
    float* caBuf = st + 1024;               // 512
    float* smaxB = st + 1536;               // 65536
    float* savgB = smaxB + 65536;           // 65536
    float* saBuf = savgB + 65536;           // 65536
    // packed f16 weights (B-fragment layout), 32B-aligned offsets
    _Float16* wp_ds  = (_Float16*)(ws + 21169664);   // 64*576   = 36864 halves
    _Float16* wp_pw1 = wp_ds  + 36864;               // 64*64    = 4096
    _Float16* wp_pw2 = wp_pw1 + 4096;                // 4096
    _Float16* wp_cbr = wp_pw2 + 4096;                // 128*1152 = 147456
    float* outF = (float*)d_out;            // residual lives here, scaled in place

    // --- pack all conv weights to f16 B-fragment layout ---
    pack_w_kernel<<<(64 * 576 + 255) / 256, 256, 0, stream>>>(ds_w,    wp_ds,  64, 576);
    pack_w_kernel<<<(64 * 64  + 255) / 256, 256, 0, stream>>>(p1_pw_w, wp_pw1, 64, 64);
    pack_w_kernel<<<(64 * 64  + 255) / 256, 256, 0, stream>>>(p2_pw_w, wp_pw2, 64, 64);
    pack_w_kernel<<<(128 * 1152 + 255) / 256, 256, 0, stream>>>(cbr_w, wp_cbr, 128, 1152);

    // --- downsample: stride-2 3x3 conv 64->64 (+BN+ReLU) into ch 0..63 ---
    conv_wmma_kernel<64, 256, 256, 3, 3, 2, 1, 1>
        <<<dim3(16384 / 16, 4), 32, 0, stream>>>(
            x, wp_ds, /*bias*/nullptr, ds_g, ds_b, buf0, 128, 0);
    // --- pools into ch 64..127 ---
    pool_kernel<<<(4 * 64 * 16384) / 256, 256, 0, stream>>>(x, buf0);

    // --- basic_conv x2 ---
    run_basic(buf0, buf2, d_in + 4,  buf1, wp_pw1, stream);   // p1: buf0 -> buf2
    run_basic(buf2, buf0, d_in + 12, buf1, wp_pw2, stream);   // p2: buf2 -> buf0

    // --- cbr: 3x3 conv 128->128 + bias + BN + ReLU -> residual in d_out ---
    conv_wmma_kernel<128, 128, 128, 3, 3, 1, 1, 2>
        <<<dim3((16384 / 16) * 2, 4), 32, 0, stream>>>(
            buf0, wp_cbr, cbr_b, cbr_g, cbr_beta, outF, 128, 0);

    // --- CSAM attention + residual (out = residual*(1 + ca*sa)) ---
    chanstat_kernel<<<512, 256, 0, stream>>>(outF, chAvg, chMax);
    ca_kernel<<<2, 256, 0, stream>>>(chAvg, chMax, ca_w, ca_b, caBuf);
    sstat_kernel<<<(4 * 16384) / 256, 256, 0, stream>>>(outF, caBuf, smaxB, savgB);
    sa_kernel<<<(4 * 16384) / 256, 256, 0, stream>>>(smaxB, savgB, sa_w, sa_b, saBuf);
    final_kernel<<<(4 * 128 * 16384) / 256, 256, 0, stream>>>(outF, caBuf, saBuf);
}
